// Qwen3MoERouter_37933151158602
// MI455X (gfx1250) — matrix operand order repaired and checked by reading, not compiled, not run on hardware
//
#include <hip/hip_runtime.h>
#include <stdint.h>

typedef __attribute__((ext_vector_type(4)))  __bf16 v4bf;
typedef __attribute__((ext_vector_type(8)))  __bf16 v8bf;
typedef __attribute__((ext_vector_type(16))) __bf16 v16bf;
typedef __attribute__((ext_vector_type(8)))  float  v8f;
typedef int v4i __attribute__((vector_size(4 * sizeof(int))));   // matches async builtin param

#define TOKENS   16384
#define HIDDEN   2048
#define EXPERTS  128
#define TOPK     8
#define TILE_T   128              // tokens per workgroup
#define KC       32               // K per chunk (one WMMA K-step), double buffered
#define NCH      (HIDDEN / KC)    // 64 chunks (even)
#define LDK      40               // bf16 row stride: 80B, 16B aligned, bank-optimal
#define LDL      (EXPERTS + 4)    // fp32 logits tile stride
#define MT       2                // M tiles per wave
#define NT       4                // N tiles per wave

#if __has_builtin(__builtin_amdgcn_global_load_async_to_lds_b128) && \
    __has_builtin(__builtin_amdgcn_s_wait_asynccnt)
#define USE_ASYNC_LDS 1
#else
#define USE_ASYNC_LDS 0
#endif

#define AS1 __attribute__((address_space(1)))
#define AS3 __attribute__((address_space(3)))

struct bf2 { __bf16 h; __bf16 l; };

__device__ __forceinline__ bf2 split2(float x) {
    bf2 r;
    r.h = (__bf16)x;                 // RNE
    r.l = (__bf16)(x - (float)r.h);  // residual
    return r;
}

__device__ __forceinline__ v16bf cat8(v8bf a, v8bf b) {
    return __builtin_shufflevector(a, b, 0,1,2,3,4,5,6,7,8,9,10,11,12,13,14,15);
}

// ---- one-shot: split gate [128,2048] fp32 -> bf16 hi/lo in workspace ----
__global__ __launch_bounds__(256)
void gate_split_kernel(const float* __restrict__ gate,
                       __bf16* __restrict__ g_hi, __bf16* __restrict__ g_lo)
{
    const int base = (blockIdx.x * 256 + threadIdx.x) * 4;
#pragma unroll
    for (int i = 0; i < 4; ++i) {
        const int idx = base + i;
        if (idx < EXPERTS * HIDDEN) {
            const bf2 r = split2(gate[idx]);
            g_hi[idx] = r.h;
            g_lo[idx] = r.l;
        }
    }
}

struct Tiles {
    __bf16 h_hi[TILE_T * LDK];      // 10,240 B each
    __bf16 h_lo[TILE_T * LDK];
    __bf16 g_hi[EXPERTS * LDK];
    __bf16 g_lo[EXPERTS * LDK];
};                                   // 40,960 B

__global__ __launch_bounds__(256)
void moe_router_kernel(const float* __restrict__ hidden,
                       const __bf16* __restrict__ g_hi,
                       const __bf16* __restrict__ g_lo,
                       float* __restrict__ topk_w,
                       float* __restrict__ topk_i,
                       float* __restrict__ logits_out)
{
    __shared__ union {
        Tiles t[2];                   // 81,920 B (double buffer)
        float logits[TILE_T * LDL];   // 67,584 B (reused after GEMM)
    } sh;

    const int tok0 = blockIdx.x * TILE_T;
    const int tid  = threadIdx.x;
    const int wave = tid >> 5;
    const int lane = tid & 31;
    const int half = lane >> 4;
    const int r16  = lane & 15;
    const int wr   = wave & 3;        // M row of wave (4 rows x MT x 16 = 128 tokens)
    const int wc   = wave >> 2;       // N col of wave (2 cols x NT x 16 = 128 experts)

    v8f acc[MT][NT] = {};

    // ---- stage chunk k0 into buffer BUF (compile-time): async gate DMA first, then hidden ----
    auto stage = [&](auto bufc, int k0) {
        constexpr int BUF = decltype(bufc)::value;
        Tiles& T = sh.t[BUF];
        // gate tile: pure copy, no VGPR data -> start the DMA engine first
#if USE_ASYNC_LDS
#pragma unroll
        for (int i = 0; i < 2; ++i) {
            const int j   = tid + i * 256;
            const int row = j >> 2;
            const int sg  = (j & 3) << 3;   // bf16 elements, 8 per 16B transfer
            __builtin_amdgcn_global_load_async_to_lds_b128(
                (AS1 v4i*)(g_hi + (size_t)row * HIDDEN + k0 + sg),
                (AS3 v4i*)&T.g_hi[row * LDK + sg], 0, 0);
            __builtin_amdgcn_global_load_async_to_lds_b128(
                (AS1 v4i*)(g_lo + (size_t)row * HIDDEN + k0 + sg),
                (AS3 v4i*)&T.g_lo[row * LDK + sg], 0, 0);
        }
#else
#pragma unroll
        for (int i = 0; i < 2; ++i) {
            const int j   = tid + i * 256;
            const int row = j >> 2;
            const int sg  = (j & 3) << 3;
            const uint4 vh = *(const uint4*)(g_hi + (size_t)row * HIDDEN + k0 + sg);
            const uint4 vl = *(const uint4*)(g_lo + (size_t)row * HIDDEN + k0 + sg);
            *(uint4*)&T.g_hi[row * LDK + sg] = vh;
            *(uint4*)&T.g_lo[row * LDK + sg] = vl;
        }
#endif
        // hidden tile: 128 rows x 32 fp32 = 1024 float4, 4 per thread (load+split+ds_store)
#pragma unroll
        for (int i = 0; i < 4; ++i) {
            const int j   = tid + i * 256;
            const int row = j >> 3;
            const int c4  = (j & 7) << 2;
            const float4 v = *(const float4*)&hidden[(size_t)(tok0 + row) * HIDDEN + k0 + c4];
            v4bf hv, lv;
            const bf2 r0 = split2(v.x);
            const bf2 r1 = split2(v.y);
            const bf2 r2 = split2(v.z);
            const bf2 r3 = split2(v.w);
            hv[0] = r0.h; lv[0] = r0.l;
            hv[1] = r1.h; lv[1] = r1.l;
            hv[2] = r2.h; lv[2] = r2.l;
            hv[3] = r3.h; lv[3] = r3.l;
            *(v4bf*)&T.h_hi[row * LDK + c4] = hv;
            *(v4bf*)&T.h_lo[row * LDK + c4] = lv;
        }
    };

    // ---- one K=32 step of bf16x2 WMMA on buffer BUF (compile-time) ----
    auto compute = [&](auto bufc) {
        constexpr int BUF = decltype(bufc)::value;
        Tiles& T = sh.t[BUF];
        v16bf Ah[MT], Al[MT], Bh[NT], Bl[NT];
#pragma unroll
        for (int mt = 0; mt < MT; ++mt) {
            // A lane layout: row (lane&15), K run [16h, 16h+16) contiguous
            const int ro = (wr * (MT * 16) + mt * 16 + r16) * LDK + 16 * half;
            Ah[mt] = cat8(*(const v8bf*)&T.h_hi[ro], *(const v8bf*)&T.h_hi[ro + 8]);
            Al[mt] = cat8(*(const v8bf*)&T.h_lo[ro], *(const v8bf*)&T.h_lo[ro + 8]);
        }
#pragma unroll
        for (int nt = 0; nt < NT; ++nt) {
            // B lane layout: col (lane&15), K run [16h, 16h+16) contiguous
            const int ro = (wc * (NT * 16) + nt * 16 + r16) * LDK + 16 * half;
            Bh[nt] = cat8(*(const v8bf*)&T.g_hi[ro], *(const v8bf*)&T.g_hi[ro + 8]);
            Bl[nt] = cat8(*(const v8bf*)&T.g_lo[ro], *(const v8bf*)&T.g_lo[ro + 8]);
        }
        // product-major order: each acc reused at distance 8 (hides XDL latency)
#pragma unroll
        for (int nt = 0; nt < NT; ++nt)
#pragma unroll
            for (int mt = 0; mt < MT; ++mt)
                acc[mt][nt] = __builtin_amdgcn_wmma_f32_16x16x32_bf16(
                    false, Ah[mt], false, Bh[nt], (short)0, acc[mt][nt], false, false);
#pragma unroll
        for (int nt = 0; nt < NT; ++nt)
#pragma unroll
            for (int mt = 0; mt < MT; ++mt)
                acc[mt][nt] = __builtin_amdgcn_wmma_f32_16x16x32_bf16(
                    false, Ah[mt], false, Bl[nt], (short)0, acc[mt][nt], false, false);
#pragma unroll
        for (int nt = 0; nt < NT; ++nt)
#pragma unroll
            for (int mt = 0; mt < MT; ++mt)
                acc[mt][nt] = __builtin_amdgcn_wmma_f32_16x16x32_bf16(
                    false, Al[mt], false, Bh[nt], (short)0, acc[mt][nt], false, false);
    };

    auto fence = [&]() {
#if USE_ASYNC_LDS
        __builtin_amdgcn_s_wait_asynccnt(0);   // async writes to the other buffer landed
#endif
        __syncthreads();                        // all reads of current buffer done
    };

    std::integral_constant<int, 0> B0;
    std::integral_constant<int, 1> B1;

    // ---- software-pipelined main loop, buffers specialized at compile time ----
    stage(B0, 0);
    fence();
    for (int k = 0; k + 1 < NCH; k += 2) {
        stage(B1, (k + 1) * KC);    // prefetch odd chunk while computing even
        compute(B0);
        fence();
        if (k + 2 < NCH) stage(B0, (k + 2) * KC);
        compute(B1);
        fence();
    }

    // ---- write logits: global + LDS tile (union reuse is safe: barrier above) ----
#pragma unroll
    for (int mt = 0; mt < MT; ++mt)
#pragma unroll
        for (int nt = 0; nt < NT; ++nt) {
            const int e = wc * (NT * 16) + nt * 16 + r16;
#pragma unroll
            for (int j = 0; j < 8; ++j) {
                const int m = wr * (MT * 16) + mt * 16 + j + 8 * half;
                const float v = acc[mt][nt][j];
                logits_out[(size_t)(tok0 + m) * EXPERTS + e] = v;
                sh.logits[m * LDL + e] = v;
            }
        }
    __syncthreads();

    // ---- per-token top-8 + renormalized softmax (wave handles 16 tokens) ----
    for (int t = 0; t < 16; ++t) {
        const int m = wave * 16 + t;
        float v[4];
#pragma unroll
        for (int i = 0; i < 4; ++i) v[i] = sh.logits[m * LDL + lane + 32 * i];
        unsigned act = 0xFu;

        float topv[TOPK];
        int   topi[TOPK];
#pragma unroll
        for (int r = 0; r < TOPK; ++r) {
            float bv = -3.4e38f;
            int   bi = 0x7FFF;
#pragma unroll
            for (int i = 0; i < 4; ++i) {
                if ((act >> i) & 1u) {
                    const float cv = v[i];
                    const int   ci = lane + 32 * i;
                    if (cv > bv || (cv == bv && ci < bi)) { bv = cv; bi = ci; }
                }
            }
            const unsigned ub = __float_as_uint(bv);
            unsigned key = (ub & 0x80000000u) ? ~ub : (ub | 0x80000000u);
            unsigned pay = (unsigned)(255 - (bi & 0xFF));
#pragma unroll
            for (int off = 16; off > 0; off >>= 1) {
                const unsigned okey = (unsigned)__shfl_xor((int)key, off, 32);
                const unsigned opay = (unsigned)__shfl_xor((int)pay, off, 32);
                if (okey > key || (okey == key && opay > pay)) { key = okey; pay = opay; }
            }
            const int widx = 255 - (int)(pay & 0xFFu);
            const unsigned uw = (key & 0x80000000u) ? (key & 0x7FFFFFFFu) : ~key;
            topv[r] = __uint_as_float(uw);
            topi[r] = widx;
            if ((widx & 31) == lane) act &= ~(1u << (widx >> 5));
        }

        const float mx = topv[0];
        float ev[TOPK];
        float s = 0.f;
#pragma unroll
        for (int r = 0; r < TOPK; ++r) { ev[r] = __expf(topv[r] - mx); s += ev[r]; }
        const float inv = 1.f / s;

        if (lane == 0) {
            const size_t base = (size_t)(tok0 + m) * TOPK;
#pragma unroll
            for (int r = 0; r < TOPK; ++r) {
                topk_w[base + r] = ev[r] * inv;
                topk_i[base + r] = (float)topi[r];
            }
        }
    }
}

extern "C" void kernel_launch(void* const* d_in, const int* in_sizes, int n_in,
                              void* d_out, int out_size, void* d_ws, size_t ws_size,
                              hipStream_t stream)
{
    const float* hidden = (const float*)d_in[0];   // [16384, 2048]
    const float* gate   = (const float*)d_in[1];   // [128, 2048]

    float* out    = (float*)d_out;
    float* topk_w = out;                               // [16384, 8]
    float* topk_i = out + (size_t)TOKENS * TOPK;       // [16384, 8] (indices as float)
    float* logits = out + (size_t)TOKENS * TOPK * 2;   // [16384, 128]

    __bf16* g_hi = (__bf16*)d_ws;                      // 512 KB
    __bf16* g_lo = g_hi + (size_t)EXPERTS * HIDDEN;    // 512 KB

    // one-shot gate split (re-run every call: deterministic)
    hipLaunchKernelGGL(gate_split_kernel, dim3(EXPERTS * HIDDEN / (256 * 4)), dim3(256),
                       0, stream, gate, g_hi, g_lo);

    hipLaunchKernelGGL(moe_router_kernel, dim3(TOKENS / TILE_T), dim3(256), 0, stream,
                       hidden, g_hi, g_lo, topk_w, topk_i, logits);
}
